// SplitQKVAttention_70317204570414
// MI455X (gfx1250) — compile-verified, matrix operand order checked by reading, not run on hardware
//
#include <hip/hip_runtime.h>
#include <hip/hip_bf16.h>
#include <stdint.h>

typedef __bf16 bf16;
typedef __attribute__((ext_vector_type(16))) __bf16 v16bf;
typedef __attribute__((ext_vector_type(8)))  float  v8f;
typedef __attribute__((ext_vector_type(4)))  int    v4i;

#define T_SEQ  2048
#define BATCH  4
#define NHEAD  16
#define HDIM   64
#define DMODEL 1024
#define KT     64

#if defined(__has_builtin)
#if __has_builtin(__builtin_amdgcn_global_load_async_to_lds_b128) && \
    __has_builtin(__builtin_amdgcn_s_wait_asynccnt)
#define HAVE_ASYNC_LDS 1
#endif
#endif
#ifndef HAVE_ASYNC_LDS
#define HAVE_ASYNC_LDS 0
#endif

#if HAVE_ASYNC_LDS
typedef __attribute__((address_space(1))) v4i* gv4i_p;  // global AS pointer
typedef __attribute__((address_space(3))) v4i* lv4i_p;  // LDS AS pointer (32-bit)
#endif

union Frag16 { v16bf v; uint4 u[2]; };
union Pack8  { bf16 h[8]; uint4 u; };

__device__ __forceinline__ v8f zero8() {
  v8f z;
#pragma unroll
  for (int i = 0; i < 8; ++i) z[i] = 0.0f;
  return z;
}

__device__ __forceinline__ v8f wmma_bf16(v16bf a, v16bf b, v8f c) {
  return __builtin_amdgcn_wmma_f32_16x16x32_bf16(false, a, false, b, (short)0, c, false, false);
}

// 16B async global->LDS copy (gfx1250 GLOBAL_LOAD_ASYNC_TO_LDS_B128).
// Global: flat address == AS1 address (64-bit).  LDS: flat->local is a
// truncation to addr[31:0] per ISA 10.2, so the low 32 bits are the LDS addr.
__device__ __forceinline__ void async_cp16(const bf16* g, bf16* l) {
#if HAVE_ASYNC_LDS
  gv4i_p gp = (gv4i_p)(unsigned long long)(uintptr_t)g;
  lv4i_p lp = (lv4i_p)(unsigned int)(uintptr_t)l;
  __builtin_amdgcn_global_load_async_to_lds_b128(gp, lp, 0, 0);
#else
  *(uint4*)l = *(const uint4*)g;
#endif
}

__device__ __forceinline__ void wait_async0() {
#if HAVE_ASYNC_LDS
  __builtin_amdgcn_s_wait_asynccnt(0);
#endif
}

// Load one 16-bit A/B fragment per ISA layout (7.12.2): per lane, two
// contiguous 8-element (16B) runs at k = khalf*8 and k = 16 + khalf*8.
__device__ __forceinline__ v16bf ldFrag(const bf16* base, int rowsel, int stride, int half) {
  Frag16 f;
  const bf16* p = base + rowsel * stride + half * 8;
  f.u[0] = *(const uint4*)(p);
  f.u[1] = *(const uint4*)(p + 16);
  return f.v;
}

__device__ __forceinline__ float half16_max(float v) {
#pragma unroll
  for (int m = 1; m < 16; m <<= 1) v = fmaxf(v, __shfl_xor(v, m, 16));
  return v;
}
__device__ __forceinline__ float half16_sum(float v) {
#pragma unroll
  for (int m = 1; m < 16; m <<= 1) v += __shfl_xor(v, m, 16);
  return v;
}

// ---------------------------------------------------------------------------
// GEMM: C[M,N] = A[M,K] @ W[N,K]^T + bias.  M=8192, N=K=1024.
// Block = 128 threads (4 waves), tile 128(M) x 64(N), K staged 32-wide,
// double-buffered LDS; each wave computes 32x64 (8 WMMAs per K-chunk,
// B fragments reused across both M-subtiles).
// ---------------------------------------------------------------------------
template <bool A_IS_F32, bool OUT_QKV>
__global__ __launch_bounds__(128) void gemm_kernel(
    const float* __restrict__ Af, const bf16* __restrict__ Ab,
    const float* __restrict__ W, const float* __restrict__ bias,
    bf16* __restrict__ outq, float* __restrict__ outf) {
  __shared__ bf16 Ash[2][128 * 32];
  __shared__ bf16 Bsh[2][64 * 32];
  const int K = DMODEL, N = DMODEL;

  const int n0 = blockIdx.x * 64;
  const int m0 = blockIdx.y * 128;
  const int tid  = threadIdx.x;
  const int wave = tid >> 5;
  const int lane = tid & 31;
  const int half = lane >> 4;
  const int l15  = lane & 15;

  v8f acc[2][4];
#pragma unroll
  for (int s = 0; s < 2; ++s)
#pragma unroll
    for (int i = 0; i < 4; ++i) acc[s][i] = zero8();

  const int arow  = tid;             // A: one row of 32 per thread
  const int brow  = tid & 63;        // B: two threads per row
  const int bpart = (tid >> 6) * 16;

  auto stage = [&](int buf, int kc) {
    if constexpr (A_IS_F32) {
      const float* src = Af + (size_t)(m0 + arow) * K + kc;
#pragma unroll
      for (int c = 0; c < 4; ++c) {
        Pack8 p;
#pragma unroll
        for (int j = 0; j < 8; ++j) p.h[j] = (bf16)src[c * 8 + j];
        *(uint4*)(&Ash[buf][arow * 32 + c * 8]) = p.u;
      }
      if (kc + 32 < K) __builtin_prefetch(src + 32, 0, 1);  // global_prefetch_b8
    } else {
      const bf16* src = Ab + (size_t)(m0 + arow) * K + kc;
#pragma unroll
      for (int c = 0; c < 4; ++c)
        async_cp16(src + c * 8, &Ash[buf][arow * 32 + c * 8]);
    }
    {
      const float* src = W + (size_t)(n0 + brow) * K + kc + bpart;
      Pack8 p0, p1;
#pragma unroll
      for (int j = 0; j < 8; ++j) { p0.h[j] = (bf16)src[j]; p1.h[j] = (bf16)src[8 + j]; }
      *(uint4*)(&Bsh[buf][brow * 32 + bpart])     = p0.u;
      *(uint4*)(&Bsh[buf][brow * 32 + bpart + 8]) = p1.u;
      if (kc + 32 < K) __builtin_prefetch(src + 32, 0, 1);
    }
  };

  stage(0, 0);
  for (int kc = 0; kc < K; kc += 32) {
    const int cb = (kc >> 5) & 1;
    wait_async0();
    __syncthreads();
    if (kc + 32 < K) stage(cb ^ 1, kc + 32);

    v16bf bfr[4];
#pragma unroll
    for (int nt = 0; nt < 4; ++nt)
      bfr[nt] = ldFrag(Bsh[cb], nt * 16 + l15, 32, half);
#pragma unroll
    for (int sub = 0; sub < 2; ++sub) {
      v16bf af = ldFrag(Ash[cb], wave * 32 + sub * 16 + l15, 32, half);
#pragma unroll
      for (int nt = 0; nt < 4; ++nt)
        acc[sub][nt] = wmma_bf16(af, bfr[nt], acc[sub][nt]);
    }
  }

#pragma unroll
  for (int nt = 0; nt < 4; ++nt) {
    const int n = n0 + nt * 16 + l15;
    const float bs = bias[n];
#pragma unroll
    for (int sub = 0; sub < 2; ++sub) {
#pragma unroll
      for (int r = 0; r < 8; ++r) {
        const int m = m0 + wave * 32 + sub * 16 + r + 8 * half;
        const float v = acc[sub][nt][r] + bs;
        if constexpr (OUT_QKV) {
          const int t = m >> 2, b = m & 3;   // m = t*B + b
          const int h = n >> 6, d = n & 63;  // n = h*64 + d
          outq[(((size_t)b * NHEAD + h) * T_SEQ + t) * HDIM + d] = (bf16)v;
        } else {
          outf[(size_t)m * N + n] = v;
        }
      }
    }
  }
}

// ---------------------------------------------------------------------------
// Flash attention: block = 4 waves, each wave owns 16 query rows.
// grid = (T/64, B*H). K/V tiles of 64 keys, double-buffered in LDS;
// K tile staged with async global->LDS copies, V transposed on the fly.
// ---------------------------------------------------------------------------
__global__ __launch_bounds__(128) void attn_kernel(
    const bf16* __restrict__ Qb, const bf16* __restrict__ Kb,
    const bf16* __restrict__ Vb, const float* __restrict__ amask,
    const uint8_t* __restrict__ kpm, bf16* __restrict__ Ob) {
  __shared__ bf16 Ksh[2][KT * HDIM];   // [key][d]
  __shared__ bf16 Vt[2][HDIM * KT];    // [d][key]
  __shared__ bf16 Psh[4 * 16 * KT];    // per-wave P tiles [q][key]

  const int bh = blockIdx.y;
  const int bb = bh >> 4;   // batch
  const int hh = bh & 15;   // head
  const int q0 = blockIdx.x * 64;
  const int tid  = threadIdx.x;
  const int wave = tid >> 5;
  const int lane = tid & 31;
  const int half = lane >> 4;
  const int l15  = lane & 15;
  const int qw   = q0 + wave * 16;

  // Q fragments held in registers for the whole K loop (2 d-chunks of 32)
  const bf16* Qrow = Qb + ((size_t)bh * T_SEQ + qw + l15) * HDIM;
  v16bf qf[2];
#pragma unroll
  for (int c = 0; c < 2; ++c) {
    Frag16 f;
    f.u[0] = *(const uint4*)(Qrow + c * 32 + half * 8);
    f.u[1] = *(const uint4*)(Qrow + c * 32 + half * 8 + 16);
    qf[c] = f.v;
  }

  v8f oacc[4];
#pragma unroll
  for (int i = 0; i < 4; ++i) oacc[i] = zero8();
  float mrun[8], lrun[8];
#pragma unroll
  for (int r = 0; r < 8; ++r) { mrun[r] = -INFINITY; lrun[r] = 0.0f; }

  const int ldrow = tid >> 1;        // key index in tile, 0..63
  const int ldp   = (tid & 1) * 32;  // d part, 0 or 32
  bf16* Pw = Psh + wave * 16 * KT;

  auto stageKV = [&](int buf, int kt0) {
    const bf16* ks = Kb + ((size_t)bh * T_SEQ + kt0 + ldrow) * HDIM + ldp;
    bf16* kd = &Ksh[buf][ldrow * HDIM + ldp];
#pragma unroll
    for (int c = 0; c < 4; ++c) async_cp16(ks + c * 8, kd + c * 8);

    const uint4* vs = (const uint4*)(Vb + ((size_t)bh * T_SEQ + kt0 + ldrow) * HDIM + ldp);
    Pack8 vp[4];
    vp[0].u = vs[0]; vp[1].u = vs[1]; vp[2].u = vs[2]; vp[3].u = vs[3];
#pragma unroll
    for (int j = 0; j < 32; ++j)
      Vt[buf][(ldp + j) * KT + ldrow] = vp[j >> 3].h[j & 7];
  };

  stageKV(0, 0);
  for (int kt0 = 0; kt0 < T_SEQ; kt0 += KT) {
    const int cb = (kt0 >> 6) & 1;
    wait_async0();
    __syncthreads();
    if (kt0 + KT < T_SEQ) stageKV(cb ^ 1, kt0 + KT);

    // S = Q K^T for this tile: 4 key-subtiles x 2 d-chunks = 8 WMMAs
    v8f sf[4];
#pragma unroll
    for (int nt = 0; nt < 4; ++nt) {
      v8f c = zero8();
#pragma unroll
      for (int ck = 0; ck < 2; ++ck) {
        v16bf bfg = ldFrag(&Ksh[cb][ck * 32], nt * 16 + l15, HDIM, half);
        c = wmma_bf16(qf[ck], bfg, c);
      }
      sf[nt] = c;
    }

    // scale + attn_mask + key_padding_mask (column fixed per lane per subtile)
#pragma unroll
    for (int nt = 0; nt < 4; ++nt) {
      const int kcol = kt0 + nt * 16 + l15;
      const bool pad = kpm[(size_t)bb * T_SEQ + kcol] != 0;
#pragma unroll
      for (int r = 0; r < 8; ++r) {
        const int q = qw + r + 8 * half;
        const float v = sf[nt][r] * 0.125f + amask[(size_t)q * T_SEQ + kcol];
        sf[nt][r] = pad ? -1e30f : v;
      }
    }

    // online softmax: per-row max across the 16 lanes of this half-wave
    float nm[8], corr[8];
#pragma unroll
    for (int r = 0; r < 8; ++r) {
      float mx = fmaxf(fmaxf(sf[0][r], sf[1][r]), fmaxf(sf[2][r], sf[3][r]));
      mx = half16_max(mx);
      nm[r] = fmaxf(mrun[r], mx);
      corr[r] = __expf(mrun[r] - nm[r]);
      mrun[r] = nm[r];
      lrun[r] *= corr[r];
    }
#pragma unroll
    for (int dt = 0; dt < 4; ++dt)
#pragma unroll
      for (int r = 0; r < 8; ++r) oacc[dt][r] *= corr[r];

    // P = exp(S - m), pack bf16 into wave-private LDS, accumulate row sums
#pragma unroll
    for (int r = 0; r < 8; ++r) {
      float rs = 0.0f;
#pragma unroll
      for (int nt = 0; nt < 4; ++nt) {
        const float e = __expf(sf[nt][r] - nm[r]);
        rs += e;
        Pw[(r + 8 * half) * KT + nt * 16 + l15] = (bf16)e;
      }
      lrun[r] += half16_sum(rs);
    }
    // cross-lane LDS RAW inside one wave: hw keeps DS in order, force the wait
    __asm__ volatile("s_wait_dscnt 0" ::: "memory");

    // O += P @ V: 2 k-chunks x 4 d-subtiles = 8 WMMAs
#pragma unroll
    for (int ck = 0; ck < 2; ++ck) {
      v16bf af = ldFrag(Pw + ck * 32, l15, KT, half);
#pragma unroll
      for (int dt = 0; dt < 4; ++dt) {
        v16bf bfg = ldFrag(&Vt[cb][ck * 32], dt * 16 + l15, KT, half);
        oacc[dt] = wmma_bf16(af, bfg, oacc[dt]);
      }
    }
  }

  // epilogue: O / l, store bf16 in (t*B+b, h*64+d) layout for the final GEMM
#pragma unroll
  for (int dt = 0; dt < 4; ++dt) {
#pragma unroll
    for (int r = 0; r < 8; ++r) {
      const int q = qw + r + 8 * half;
      const int d = dt * 16 + l15;
      const float o = oacc[dt][r] / lrun[r];
      Ob[((size_t)q * BATCH + bb) * DMODEL + hh * HDIM + d] = (bf16)o;
    }
  }
}

// ---------------------------------------------------------------------------
extern "C" void kernel_launch(void* const* d_in, const int* in_sizes, int n_in,
                              void* d_out, int out_size, void* d_ws, size_t ws_size,
                              hipStream_t stream) {
  const float*   query = (const float*)d_in[0];
  const float*   key   = (const float*)d_in[1];
  const float*   value = (const float*)d_in[2];
  const uint8_t* kpm   = (const uint8_t*)d_in[3];
  const float*   amask = (const float*)d_in[4];
  const float* Wq = (const float*)d_in[5];
  const float* bq = (const float*)d_in[6];
  const float* Wk = (const float*)d_in[7];
  const float* bk = (const float*)d_in[8];
  const float* Wv = (const float*)d_in[9];
  const float* bv = (const float*)d_in[10];
  const float* Wo = (const float*)d_in[11];
  const float* bo = (const float*)d_in[12];
  float* out = (float*)d_out;

  const size_t elems = (size_t)T_SEQ * BATCH * DMODEL;  // 8M
  bf16* Qb = (bf16*)d_ws;        // 16 MB each, 64 MB total of workspace
  bf16* Kb = Qb + elems;
  bf16* Vb = Kb + elems;
  bf16* Ob = Vb + elems;

  dim3 blk(128);
  dim3 ggrid(DMODEL / 64, (T_SEQ * BATCH) / 128);

  gemm_kernel<true, true><<<ggrid, blk, 0, stream>>>(query, nullptr, Wq, bq, Qb, nullptr);
  gemm_kernel<true, true><<<ggrid, blk, 0, stream>>>(key,   nullptr, Wk, bk, Kb, nullptr);
  gemm_kernel<true, true><<<ggrid, blk, 0, stream>>>(value, nullptr, Wv, bv, Vb, nullptr);

  dim3 agrid(T_SEQ / 64, BATCH * NHEAD);
  attn_kernel<<<agrid, blk, 0, stream>>>(Qb, Kb, Vb, amask, kpm, Ob);

  gemm_kernel<false, false><<<ggrid, blk, 0, stream>>>(nullptr, Ob, Wo, bo, nullptr, out);
}
